// DynamicsBackbone_86887188398387
// MI455X (gfx1250) — compile-verified, matrix operand order checked by reading, not run on hardware
//
#include <hip/hip_runtime.h>
#include <hip/hip_bf16.h>
#include <stdint.h>

// ---------------------------------------------------------------------------
// DynamicsBackbone for MI455X (gfx1250).
// Compute-bound (~0.75 TFLOP vs ~13us of HBM traffic at 23.3TB/s), so all
// GEMMs go through v_wmma_f32_16x16x32_bf16 (bf16 in, fp32 accumulate).
// Weights are converted fp32 -> bf16 *transposed* each call so the WMMA
// B-fragment (per-lane column of B = row of B^T) is two contiguous 16B chunks.
// The hidden dim (2730) is padded to 2752 so every GEMM operand has
// 16B-aligned rows and K % 32 == 0 -> branch-free vectorized LDS staging.
// Requires ws_size >= ~180 MiB.
// ---------------------------------------------------------------------------

typedef __attribute__((ext_vector_type(16))) __bf16 v16bf;
typedef __attribute__((ext_vector_type(8)))  float  v8f;

constexpr int B_  = 2;
constexpr int T_  = 8;
constexpr int H_  = 16;
constexpr int W_  = 16;
constexpr int C_  = 5;
constexpr int A_  = 8;
constexpr int D_  = 1024;
constexpr int NH_ = 16;
constexpr int HD_ = 64;
constexpr int GH_ = 8;
constexpr int LH_ = NH_ - GH_;    // 8
constexpr int TS_ = 8;
constexpr int L_  = 6;
constexpr int HID_  = 2730;       // int(D*4*2/3)
constexpr int HIDP_ = 2752;       // HID padded: %32==0, bf16 rows 16B aligned
constexpr int S_  = T_ * H_ * W_; // 2048
constexpr int NT_ = (H_ / TS_) * (W_ / TS_); // 4 tiles
constexpr int SL_ = T_ * TS_ * TS_;          // 512 tokens per local tile
constexpr int M_ROWS = B_ * S_;              // 4096

// ======================= generic bf16 WMMA GEMM ============================
// C[m][n] = alpha * sum_k A[m][k] * Bt[n][k]   (+ C_old if CADD)
// A: bf16 row-major (lda), Bt: bf16 "transposed B" row-major (ldb),
// C: fp32 (ldc). Batched over blockIdx.z with element strides.
// REQUIRES: K % 32 == 0; A/Bt bases and lda/ldb give 16B-aligned rows.
template <bool CADD>
__global__ __launch_bounds__(256) void gemm_bf16_kernel(
    const __bf16* __restrict__ A, long long sAz, int lda,
    const __bf16* __restrict__ Bt, long long sBz, int ldb,
    float* __restrict__ C, long long sCz, int ldc,
    int M, int N, int K, float alpha)
{
    A  += (size_t)blockIdx.z * (size_t)sAz;
    Bt += (size_t)blockIdx.z * (size_t)sBz;
    C  += (size_t)blockIdx.z * (size_t)sCz;

    const int bm = blockIdx.y * 128;
    const int bn = blockIdx.x * 128;

    __shared__ __align__(16) __bf16 sA[128][40];  // 32 K + 8 pad (80B rows)
    __shared__ __align__(16) __bf16 sB[128][40];

    const int tid  = threadIdx.x;
    const int wave = tid >> 5;
    const int lane = tid & 31;
    const int l15  = lane & 15;
    const int hf   = lane >> 4;           // lane half selects K sub-range
    const int wm0  = (wave >> 1) * 32;    // 4 waves along M (32 rows each)
    const int wn0  = (wave & 1) * 64;     // 2 waves along N (64 cols each)

    // Per-thread staging coordinates (2 chunks of 8 bf16 per matrix).
    const int row0 = tid >> 2,       kc0 = (tid & 3) * 8;
    const int row1 = (tid + 256) >> 2, kc1 = ((tid + 256) & 3) * 8;
    const bool aok0 = (bm + row0) < M, aok1 = (bm + row1) < M;
    const bool bok0 = (bn + row0) < N, bok1 = (bn + row1) < N;
    const __bf16* pa0 = A + (size_t)(bm + row0) * lda + kc0;
    const __bf16* pa1 = A + (size_t)(bm + row1) * lda + kc1;
    const __bf16* pb0 = Bt + (size_t)(bn + row0) * ldb + kc0;
    const __bf16* pb1 = Bt + (size_t)(bn + row1) * ldb + kc1;
    const uint4 zero4 = make_uint4(0u, 0u, 0u, 0u);

    v8f acc[2][4];
    for (int tm = 0; tm < 2; ++tm)
        for (int tn = 0; tn < 4; ++tn)
            for (int e = 0; e < 8; ++e) acc[tm][tn][e] = 0.0f;

    for (int k0 = 0; k0 < K; k0 += 32) {
        __syncthreads();
        // ---- stage 128x32 tiles of A and Bt (vectorized, branch-light)
        {
            uint4 va0 = aok0 ? *(const uint4*)(pa0 + k0) : zero4;
            uint4 va1 = aok1 ? *(const uint4*)(pa1 + k0) : zero4;
            uint4 vb0 = bok0 ? *(const uint4*)(pb0 + k0) : zero4;
            uint4 vb1 = bok1 ? *(const uint4*)(pb1 + k0) : zero4;
            if (k0 + 32 < K) {  // prefetch next K tile (global_prefetch_b8)
                if (aok0) __builtin_prefetch((const void*)(pa0 + k0 + 32), 0, 1);
                if (bok0) __builtin_prefetch((const void*)(pb0 + k0 + 32), 0, 1);
            }
            *(uint4*)&sA[row0][kc0] = va0;
            *(uint4*)&sA[row1][kc1] = va1;
            *(uint4*)&sB[row0][kc0] = vb0;
            *(uint4*)&sB[row1][kc1] = vb1;
        }
        __syncthreads();

        // ---- build fragments (16-bit A 16x32 layout: lane half picks K
        //      chunks [hf*8, hf*8+8) and [16+hf*8, 16+hf*8+8))
        v16bf af[2], bfr[4];
        for (int tm = 0; tm < 2; ++tm) {
            int row = wm0 + tm * 16 + l15;
            union { uint4 q[2]; v16bf v; } u;
            u.q[0] = *(const uint4*)&sA[row][hf * 8];
            u.q[1] = *(const uint4*)&sA[row][16 + hf * 8];
            af[tm] = u.v;
        }
        for (int tn = 0; tn < 4; ++tn) {
            int row = wn0 + tn * 16 + l15;
            union { uint4 q[2]; v16bf v; } u;
            u.q[0] = *(const uint4*)&sB[row][hf * 8];
            u.q[1] = *(const uint4*)&sB[row][16 + hf * 8];
            bfr[tn] = u.v;
        }
        for (int tm = 0; tm < 2; ++tm)
            for (int tn = 0; tn < 4; ++tn)
                acc[tm][tn] = __builtin_amdgcn_wmma_f32_16x16x32_bf16(
                    false, af[tm], false, bfr[tn], (short)0, acc[tm][tn],
                    false, false);
    }

    // ---- store (C/D layout: vgpr v -> M = v + 8*hf, lanes l15 -> N)
    for (int tm = 0; tm < 2; ++tm)
        for (int tn = 0; tn < 4; ++tn)
            for (int v = 0; v < 8; ++v) {
                int m = bm + wm0 + tm * 16 + v + 8 * hf;
                int n = bn + wn0 + tn * 16 + l15;
                if (m < M && n < N) {
                    size_t o = (size_t)m * ldc + n;
                    float val = alpha * acc[tm][tn][v];
                    if (CADD) val += C[o];
                    C[o] = val;
                }
            }
}

static inline void launch_gemm(hipStream_t st,
                               const __bf16* A, long long sAz, int lda,
                               const __bf16* Bt, long long sBz, int ldb,
                               float* C, long long sCz, int ldc,
                               int M, int N, int K, float alpha, bool cadd,
                               int Z)
{
    dim3 grid((N + 127) / 128, (M + 127) / 128, Z), blk(256);
    if (cadd)
        gemm_bf16_kernel<true><<<grid, blk, 0, st>>>(A, sAz, lda, Bt, sBz, ldb,
                                                     C, sCz, ldc, M, N, K, alpha);
    else
        gemm_bf16_kernel<false><<<grid, blk, 0, st>>>(A, sAz, lda, Bt, sBz, ldb,
                                                      C, sCz, ldc, M, N, K, alpha);
}

// ===================== elementwise / helper kernels ========================

// fp32 KxN (row-major) -> bf16 NxK' (row-major, transposed, K padded to ldt;
// pad region [K, ldt) is zero-filled).  grid: (ceil(N/32), ceil(ldt/32))
__global__ void transpose_bf16_kernel(const float* __restrict__ Wsrc,
                                      __bf16* __restrict__ Wt, int K, int N,
                                      int ldt)
{
    __shared__ float tile[32][33];
    int nb = blockIdx.x * 32, kb = blockIdx.y * 32;
    int tx = threadIdx.x, ty = threadIdx.y; // 32 x 8
    for (int i = ty; i < 32; i += 8) {
        int k = kb + i, n = nb + tx;
        tile[i][tx] = (k < K && n < N) ? Wsrc[(size_t)k * N + n] : 0.0f;
    }
    __syncthreads();
    for (int i = ty; i < 32; i += 8) {
        int n = nb + i, k = kb + tx;
        if (n < N && k < ldt) Wt[(size_t)n * ldt + k] = (__bf16)tile[tx][i];
    }
}

// x = latent @ Wlp + blp + silu(action @ Wae + bae)   (tiny K: 5 and 8)
__global__ void embed_kernel(const float* __restrict__ lat,
                             const float* __restrict__ act,
                             const float* __restrict__ Wlp,
                             const float* __restrict__ blp,
                             const float* __restrict__ Wae,
                             const float* __restrict__ bae,
                             float* __restrict__ x)
{
    size_t idx = (size_t)blockIdx.x * 256 + threadIdx.x; // over B*S*D
    int d = idx & (D_ - 1);
    int s = (int)(idx >> 10) & (S_ - 1);
    int b = (int)(idx >> 21);
    int t = s >> 8, h = (s >> 4) & 15, w = s & 15;
    float a = blp[d];
    const float* lp = lat + ((size_t)((b * T_ + t) * H_ + h) * W_ + w) * C_;
    for (int c = 0; c < C_; ++c) a += lp[c] * Wlp[c * D_ + d];
    float aa = bae[d];
    const float* ap = act + (size_t)(b * T_ + t) * A_;
    for (int j = 0; j < A_; ++j) aa += ap[j] * Wae[j * D_ + d];
    a += aa / (1.0f + __expf(-aa)); // silu
    x[idx] = a;
}

// per-row RMSNorm -> bf16
__global__ void rmsnorm_bf16_kernel(const float* __restrict__ x,
                                    const float* __restrict__ w,
                                    __bf16* __restrict__ out)
{
    __shared__ float red[256];
    int row = blockIdx.x;
    const float* xr = x + (size_t)row * D_;
    float ss = 0.0f;
    for (int i = threadIdx.x; i < D_; i += 256) { float v = xr[i]; ss += v * v; }
    red[threadIdx.x] = ss; __syncthreads();
    for (int o = 128; o > 0; o >>= 1) {
        if (threadIdx.x < o) red[threadIdx.x] += red[threadIdx.x + o];
        __syncthreads();
    }
    float inv = rsqrtf(red[0] / (float)D_ + 1e-6f);
    for (int i = threadIdx.x; i < D_; i += 256)
        out[(size_t)row * D_ + i] = (__bf16)(xr[i] * inv * w[i]);
}

// RoPE on q,k + scatter into global-head and local-tile bf16 layouts
__global__ void rope_scatter_kernel(const float* __restrict__ q,
                                    const float* __restrict__ k,
                                    __bf16* __restrict__ qg, __bf16* __restrict__ kg,
                                    __bf16* __restrict__ ql, __bf16* __restrict__ kl)
{
    size_t idx = (size_t)blockIdx.x * 256 + threadIdx.x; // B*S*NH*32 pairs
    int p    = idx & 31;
    int head = (int)(idx >> 5) & 15;
    int s    = (int)(idx >> 9) & (S_ - 1);
    int b    = (int)(idx >> 20);
    int t = s >> 8, h = (s >> 4) & 15, w = s & 15;
    int off, halfd, jj; float pos;
    if (p < 8)       { off = 0;  halfd = 8;  jj = p;      pos = (float)t; }
    else if (p < 20) { off = 16; halfd = 12; jj = p - 8;  pos = (float)h; }
    else             { off = 40; halfd = 12; jj = p - 20; pos = (float)w; }
    float freq = __expf(-(float)jj / (float)halfd * 9.210340372f); // ln(1e4)
    float ang = pos * freq;
    float c = __cosf(ang), sn = __sinf(ang);
    int d0 = off + 2 * jj;
    size_t base = ((size_t)(b * S_ + s) * NH_ + head) * HD_ + d0;
    float q1 = q[base], q2 = q[base + 1];
    float k1 = k[base], k2 = k[base + 1];
    __bf16 qo1 = (__bf16)(q1 * c - q2 * sn), qo2 = (__bf16)(q1 * sn + q2 * c);
    __bf16 ko1 = (__bf16)(k1 * c - k2 * sn), ko2 = (__bf16)(k1 * sn + k2 * c);
    if (head < GH_) {
        size_t dst = ((size_t)(b * S_ + s) * GH_ + head) * HD_ + d0;
        qg[dst] = qo1; qg[dst + 1] = qo2;
        kg[dst] = ko1; kg[dst + 1] = ko2;
    } else {
        int tile = (h >> 3) * 2 + (w >> 3);
        int sl = t * 64 + (h & 7) * 8 + (w & 7);
        size_t dst = (((size_t)(b * NT_ + tile) * SL_ + sl) * LH_ + (head - GH_)) * HD_ + d0;
        ql[dst] = qo1; ql[dst + 1] = qo2;
        kl[dst] = ko1; kl[dst + 1] = ko2;
    }
}

// V -> transposed bf16 ([head][d][token]) so PV GEMM B-fragment is contiguous
__global__ void vtrans_kernel(const float* __restrict__ v,
                              __bf16* __restrict__ vtg, __bf16* __restrict__ vtl)
{
    size_t idx = (size_t)blockIdx.x * 256 + threadIdx.x; // B*S*NH*HD
    int d    = idx & 63;
    int head = (int)(idx >> 6) & 15;
    int s    = (int)(idx >> 10) & (S_ - 1);
    int b    = (int)(idx >> 21);
    float val = v[((size_t)(b * S_ + s) * NH_ + head) * HD_ + d];
    if (head < GH_) {
        vtg[(((size_t)b * GH_ + head) * HD_ + d) * S_ + s] = (__bf16)val;
    } else {
        int t = s >> 8, h = (s >> 4) & 15, w = s & 15;
        int tile = (h >> 3) * 2 + (w >> 3);
        int sl = t * 64 + (h & 7) * 8 + (w & 7);
        vtl[(((size_t)(b * NT_ + tile) * LH_ + (head - GH_)) * HD_ + d) * SL_ + sl] =
            (__bf16)val;
    }
}

// row softmax fp32 -> bf16 probs
__global__ void softmax_kernel(const float* __restrict__ Sm,
                               __bf16* __restrict__ P, int nc)
{
    __shared__ float red[256];
    int row = blockIdx.x;
    const float* r = Sm + (size_t)row * nc;
    float m = -3.4e38f;
    for (int i = threadIdx.x; i < nc; i += 256) m = fmaxf(m, r[i]);
    red[threadIdx.x] = m; __syncthreads();
    for (int o = 128; o > 0; o >>= 1) {
        if (threadIdx.x < o) red[threadIdx.x] = fmaxf(red[threadIdx.x], red[threadIdx.x + o]);
        __syncthreads();
    }
    float mx = red[0]; __syncthreads();
    float ssum = 0.0f;
    for (int i = threadIdx.x; i < nc; i += 256) ssum += __expf(r[i] - mx);
    red[threadIdx.x] = ssum; __syncthreads();
    for (int o = 128; o > 0; o >>= 1) {
        if (threadIdx.x < o) red[threadIdx.x] += red[threadIdx.x + o];
        __syncthreads();
    }
    float inv = 1.0f / red[0];
    for (int i = threadIdx.x; i < nc; i += 256)
        P[(size_t)row * nc + i] = (__bf16)(__expf(r[i] - mx) * inv);
}

// merge global (direct) + local (tile-gathered) attention outputs -> bf16
__global__ void gather_o_kernel(const float* __restrict__ of,
                                const float* __restrict__ ol,
                                __bf16* __restrict__ obf)
{
    size_t idx = (size_t)blockIdx.x * 256 + threadIdx.x; // B*S*D
    int d    = idx & (D_ - 1);
    int s    = (int)(idx >> 10) & (S_ - 1);
    int b    = (int)(idx >> 21);
    int head = d >> 6, dd = d & 63;
    float val;
    if (head < GH_) {
        val = of[idx];
    } else {
        int t = s >> 8, h = (s >> 4) & 15, w = s & 15;
        int tile = (h >> 3) * 2 + (w >> 3);
        int sl = t * 64 + (h & 7) * 8 + (w & 7);
        val = ol[(((size_t)(b * NT_ + tile) * SL_ + sl) * LH_ + (head - GH_)) * HD_ + dd];
    }
    obf[idx] = (__bf16)val;
}

// hb[row][col] = col<HID ? bf16(silu(h)*g) : 0   (padded bf16 layout, HIDP)
// grid: (ceil(HIDP/256), M_ROWS)
__global__ void silu_mul_kernel(const float* __restrict__ h,
                                const float* __restrict__ g,
                                __bf16* __restrict__ out)
{
    int col = blockIdx.x * 256 + threadIdx.x;
    int row = blockIdx.y;
    if (col >= HIDP_) return;
    __bf16 v = (__bf16)0.0f;
    if (col < HID_) {
        float x = h[(size_t)row * HID_ + col];
        v = (__bf16)((x / (1.0f + __expf(-x))) * g[(size_t)row * HID_ + col]);
    }
    out[(size_t)row * HIDP_ + col] = v;
}

// final RMSNorm of last timestep -> fp32 d_out [B][H][W][D]
__global__ void final_norm_kernel(const float* __restrict__ x,
                                  const float* __restrict__ w,
                                  float* __restrict__ out)
{
    __shared__ float red[256];
    int row = blockIdx.x;              // B*H*W rows
    int b = row >> 8, hw = row & 255;
    const float* xr = x + ((size_t)b * S_ + (T_ - 1) * H_ * W_ + hw) * D_;
    float ss = 0.0f;
    for (int i = threadIdx.x; i < D_; i += 256) { float v = xr[i]; ss += v * v; }
    red[threadIdx.x] = ss; __syncthreads();
    for (int o = 128; o > 0; o >>= 1) {
        if (threadIdx.x < o) red[threadIdx.x] += red[threadIdx.x + o];
        __syncthreads();
    }
    float inv = rsqrtf(red[0] / (float)D_ + 1e-6f);
    for (int i = threadIdx.x; i < D_; i += 256)
        out[(size_t)row * D_ + i] = xr[i] * inv * w[i];
}

// ============================ host orchestration ===========================
extern "C" void kernel_launch(void* const* d_in, const int* in_sizes, int n_in,
                              void* d_out, int out_size, void* d_ws, size_t ws_size,
                              hipStream_t stream)
{
    const float* lat = (const float*)d_in[0];
    const float* act = (const float*)d_in[1];
    const float* Wlp = (const float*)d_in[2];
    const float* blp = (const float*)d_in[3];
    const float* Wae = (const float*)d_in[4];
    const float* bae = (const float*)d_in[5];
    const float* n1w = (const float*)d_in[6];
    const float* Wq  = (const float*)d_in[7];
    const float* Wk  = (const float*)d_in[8];
    const float* Wv  = (const float*)d_in[9];
    const float* Wo  = (const float*)d_in[10];
    const float* n2w = (const float*)d_in[11];
    const float* mw  = (const float*)d_in[12];
    const float* mv  = (const float*)d_in[13];
    const float* mo  = (const float*)d_in[14];
    const float* now = (const float*)d_in[15];

    constexpr size_t MB = 1ull << 20;
    char* ws = (char*)d_ws;
    float*  xf  = (float*)(ws + 0);          // 16MB residual stream
    __bf16* xn  = (__bf16*)(ws + 16 * MB);   // 8MB  normed activations (bf16)
    __bf16* wqT = (__bf16*)(ws + 24 * MB);   // 2MB each
    __bf16* wkT = (__bf16*)(ws + 26 * MB);
    __bf16* wvT = (__bf16*)(ws + 28 * MB);
    __bf16* woT = (__bf16*)(ws + 30 * MB);
    __bf16* mwT = (__bf16*)(ws + 32 * MB);   // [HID][D] bf16 (~5.6MB)
    __bf16* mvT = (__bf16*)(ws + 38 * MB);   // [HID][D]
    __bf16* moT = (__bf16*)(ws + 44 * MB);   // [D][HIDP] (~5.5MB)
    char* big = ws + 50 * MB;                // phase-overlapped scratch (128MB)
    // --- attention phase layout
    float*  qf  = (float*)(big);             // 16MB
    float*  kf  = (float*)(big + 16 * MB);   // 16MB
    float*  vf  = (float*)(big + 32 * MB);   // 16MB
    __bf16* qg  = (__bf16*)(big + 48 * MB);  // 4MB [b][s][GH*64]
    __bf16* kg  = (__bf16*)(big + 52 * MB);  // 4MB
    __bf16* ql  = (__bf16*)(big + 56 * MB);  // 4MB [b][tile][512][LH*64]
    __bf16* kl  = (__bf16*)(big + 60 * MB);  // 4MB
    __bf16* vtg = (__bf16*)(big + 64 * MB);  // 4MB [b][g][64][S]
    __bf16* vtl = (__bf16*)(big + 68 * MB);  // 4MB [b][tile][lh][64][512]
    float*  sc  = (float*)(big + 72 * MB);   // 16MB scores
    __bf16* pr  = (__bf16*)(big + 88 * MB);  // 8MB probs
    float*  of  = (float*)(big + 96 * MB);   // 16MB [b][s][1024] (cols 0..511 used)
    float*  olb = (float*)(big + 112 * MB);  // 8MB local attn out (tile order)
    __bf16* obf = (__bf16*)(big + 120 * MB); // 8MB merged attn out (bf16)
    // --- MLP phase layout (reuses `big`)
    float*  h1 = (float*)(big);              // 4096 x 2730 fp32 (~43MB)
    float*  g1 = (float*)(big + 46 * MB);
    __bf16* hb = (__bf16*)(big + 92 * MB);   // 4096 x HIDP bf16 (~22.5MB)

    const float scale = 0.125f; // 1/sqrt(64)
    dim3 tb(32, 8);

    // ---- input embedding
    embed_kernel<<<dim3((B_ * S_ * D_) / 256), 256, 0, stream>>>(
        lat, act, Wlp, blp, Wae, bae, xf);

    for (int i = 0; i < L_; ++i) {
        // ---- weights -> bf16 transposed (per layer, ~50MB fp32 read)
        transpose_bf16_kernel<<<dim3(D_ / 32, D_ / 32), tb, 0, stream>>>(
            Wq + (size_t)i * D_ * D_, wqT, D_, D_, D_);
        transpose_bf16_kernel<<<dim3(D_ / 32, D_ / 32), tb, 0, stream>>>(
            Wk + (size_t)i * D_ * D_, wkT, D_, D_, D_);
        transpose_bf16_kernel<<<dim3(D_ / 32, D_ / 32), tb, 0, stream>>>(
            Wv + (size_t)i * D_ * D_, wvT, D_, D_, D_);
        transpose_bf16_kernel<<<dim3(D_ / 32, D_ / 32), tb, 0, stream>>>(
            Wo + (size_t)i * D_ * D_, woT, D_, D_, D_);

        // ---- attention
        rmsnorm_bf16_kernel<<<dim3(M_ROWS), 256, 0, stream>>>(xf, n1w + i * D_, xn);
        launch_gemm(stream, xn, 0, D_, wqT, 0, D_, qf, 0, D_, M_ROWS, D_, D_, 1.f, false, 1);
        launch_gemm(stream, xn, 0, D_, wkT, 0, D_, kf, 0, D_, M_ROWS, D_, D_, 1.f, false, 1);
        launch_gemm(stream, xn, 0, D_, wvT, 0, D_, vf, 0, D_, M_ROWS, D_, D_, 1.f, false, 1);
        rope_scatter_kernel<<<dim3((B_ * S_ * NH_ * 32) / 256), 256, 0, stream>>>(
            qf, kf, qg, kg, ql, kl);
        vtrans_kernel<<<dim3((B_ * S_ * NH_ * HD_) / 256), 256, 0, stream>>>(vf, vtg, vtl);

        // global heads: full 2048x2048 attention, one (b,g) at a time
        for (int b = 0; b < B_; ++b)
            for (int g = 0; g < GH_; ++g) {
                const __bf16* qgp = qg + (size_t)b * S_ * GH_ * HD_ + g * HD_;
                const __bf16* kgp = kg + (size_t)b * S_ * GH_ * HD_ + g * HD_;
                launch_gemm(stream, qgp, 0, GH_ * HD_, kgp, 0, GH_ * HD_,
                            sc, 0, S_, S_, S_, HD_, scale, false, 1);
                softmax_kernel<<<dim3(S_), 256, 0, stream>>>(sc, pr, S_);
                launch_gemm(stream, pr, 0, S_,
                            vtg + ((size_t)b * GH_ + g) * HD_ * S_, 0, S_,
                            of + (size_t)b * S_ * D_ + g * HD_, 0, D_,
                            S_, HD_, S_, 1.f, false, 1);
            }

        // local heads: 512-token tiles, batched over LH heads via grid.z
        for (int b = 0; b < B_; ++b)
            for (int tl = 0; tl < NT_; ++tl) {
                size_t qoff = (size_t)(b * NT_ + tl) * SL_ * LH_ * HD_;
                launch_gemm(stream, ql + qoff, HD_, LH_ * HD_,
                            kl + qoff, HD_, LH_ * HD_,
                            sc, (long long)SL_ * SL_, SL_,
                            SL_, SL_, HD_, scale, false, LH_);
                softmax_kernel<<<dim3(LH_ * SL_), 256, 0, stream>>>(sc, pr, SL_);
                launch_gemm(stream, pr, (long long)SL_ * SL_, SL_,
                            vtl + (size_t)(b * NT_ + tl) * LH_ * HD_ * SL_,
                            (long long)HD_ * SL_, SL_,
                            olb + qoff, HD_, LH_ * HD_,
                            SL_, HD_, SL_, 1.f, false, LH_);
            }

        gather_o_kernel<<<dim3((B_ * S_ * D_) / 256), 256, 0, stream>>>(of, olb, obf);
        // x += attn_out @ Wo  (fused residual via CADD)
        launch_gemm(stream, obf, 0, D_, woT, 0, D_, xf, 0, D_, M_ROWS, D_, D_, 1.f, true, 1);

        // ---- SwiGLU MLP (big scratch now reused for h/g/hb)
        transpose_bf16_kernel<<<dim3((HID_ + 31) / 32, D_ / 32), tb, 0, stream>>>(
            mw + (size_t)i * D_ * HID_, mwT, D_, HID_, D_);
        transpose_bf16_kernel<<<dim3((HID_ + 31) / 32, D_ / 32), tb, 0, stream>>>(
            mv + (size_t)i * D_ * HID_, mvT, D_, HID_, D_);
        transpose_bf16_kernel<<<dim3(D_ / 32, HIDP_ / 32), tb, 0, stream>>>(
            mo + (size_t)i * HID_ * D_, moT, HID_, D_, HIDP_);

        rmsnorm_bf16_kernel<<<dim3(M_ROWS), 256, 0, stream>>>(xf, n2w + i * D_, xn);
        launch_gemm(stream, xn, 0, D_, mwT, 0, D_, h1, 0, HID_, M_ROWS, HID_, D_, 1.f, false, 1);
        launch_gemm(stream, xn, 0, D_, mvT, 0, D_, g1, 0, HID_, M_ROWS, HID_, D_, 1.f, false, 1);
        silu_mul_kernel<<<dim3((HIDP_ + 255) / 256, M_ROWS), 256, 0, stream>>>(h1, g1, hb);
        // x += (silu(h)*g) @ mlp_o  (fused residual, padded K contributes 0)
        launch_gemm(stream, hb, 0, HIDP_, moT, 0, HIDP_, xf, 0, D_, M_ROWS, D_, HIDP_, 1.f, true, 1);
    }

    // ---- final norm of last timestep -> d_out (fp32, B*H*W*D)
    final_norm_kernel<<<dim3(B_ * H_ * W_), 256, 0, stream>>>(xf, now, (float*)d_out);
}
